// DCNV2_59150289601022
// MI455X (gfx1250) — matrix operand order repaired and checked by reading, not compiled, not run on hardware
//
#include <hip/hip_runtime.h>
#include <hip/hip_bf16.h>
#include <math.h>

typedef __attribute__((ext_vector_type(16))) __bf16 v16bf;
typedef __attribute__((ext_vector_type(8)))  float  v8f;
typedef __attribute__((ext_vector_type(4)))  float  v4f;

#define B_  8
#define C_  64
#define H_  128
#define W_  128
#define CO_ 64
#define KK_ 9
#define HO_ 128
#define WO_ 128
#define NSTEP 18          // K = KK_*C_ = 576 = 18 * 32

// ---------------- workspace layout (bytes) ----------------
// xt    : B*H*W*C fp32 (NHWC)                    = 33,554,432
// pkOM  : 18*2*32*16 bf16 packed B-frags          =     36,864
// pkW   : 18*4*32*16 bf16 packed B-frags          =     73,728
// dyA/dxA/mA : B*KK*HO*WO fp32 each               =  4,718,592 each
#define OFF_XT    0
#define OFF_PKOM  33554432UL
#define OFF_PKW   33591296UL
#define OFF_DY    33665024UL
#define OFF_DX    38383616UL
#define OFF_M     43102208UL

#define PKW_ELEMS  (NSTEP * 4 * 32 * 16)   // 36864 bf16 = 73728 B
#define PKOM_ELEMS (NSTEP * 2 * 32 * 16)   // 18432 bf16 = 36864 B

// Async copy of 16-byte chunks from global to LDS (GVS addressing),
// tracked with ASYNCcnt.  One chunk per lane per issue.
__device__ __forceinline__ void async_g2l_b128(unsigned lds_addr, unsigned goff,
                                               const void* sbase) {
  asm volatile("global_load_async_to_lds_b128 %0, %1, %2"
               :
               : "v"(lds_addr), "v"(goff), "s"(sbase)
               : "memory");
}
__device__ __forceinline__ void wait_async0() {
  asm volatile("s_wait_asynccnt 0x0" ::: "memory");
}

// ---------------- prep: NCHW fp32 -> NHWC fp32 ----------------
__global__ void k_xt(const float* __restrict__ x, float* __restrict__ xt) {
  int idx = blockIdx.x * blockDim.x + threadIdx.x;   // flat NCHW
  int xx = idx & (W_ - 1);
  int y  = (idx >> 7)  & (H_ - 1);
  int c  = (idx >> 14) & (C_ - 1);
  int b  = idx >> 20;
  xt[(((b * H_ + y) * W_ + xx) << 6) + c] = x[idx];
}

// ---------------- prep: pack w_om into B-fragment order ----------------
// packed index = ((s*2 + n)*32 + lane)*16 + i ;
// value = B[k = 32s + (lane>>4)*16 + i][ch = n*16 + (lane&15)]
__global__ void k_pack_om(const float* __restrict__ w_om, __bf16* __restrict__ pk) {
  int t = blockIdx.x * blockDim.x + threadIdx.x;
  if (t >= PKOM_ELEMS) return;
  int i = t & 15;
  int l = (t >> 4) & 31;
  int n = (t >> 9) & 1;
  int s = t >> 10;
  int k  = 32 * s + ((l >> 4) * 16) + i;   // k = kk*64 + c
  int kk = k >> 6;
  int c  = k & 63;
  int ch = n * 16 + (l & 15);
  float v = 0.f;
  if (ch < 27) v = w_om[(ch * C_ + c) * KK_ + kk];
  pk[t] = (__bf16)v;
}

// ---------------- prep: pack w into B-fragment order (4 N-chunks) ----------------
__global__ void k_pack_w(const float* __restrict__ w, __bf16* __restrict__ pk) {
  int t = blockIdx.x * blockDim.x + threadIdx.x;
  if (t >= PKW_ELEMS) return;
  int i = t & 15;
  int l = (t >> 4) & 31;
  int n = (t >> 9) & 3;
  int s = t >> 11;
  int k  = 32 * s + ((l >> 4) * 16) + i;
  int kk = k >> 6;
  int c  = k & 63;
  int co = n * 16 + (l & 15);
  pk[t] = (__bf16)w[(co * C_ + c) * KK_ + kk];
}

// ---------------- kernel 2: offset/mask conv via WMMA ----------------
__global__ __launch_bounds__(128) void k_offset(
    const float* __restrict__ xt, const __bf16* __restrict__ pk,
    const float* __restrict__ b_om,
    float* __restrict__ dyA, float* __restrict__ dxA, float* __restrict__ mA) {
  __shared__ __bf16 ldsB[PKOM_ELEMS];

  const int tid  = threadIdx.x;
  const int lane = tid & 31;
  const int wave = tid >> 5;
  const int hlf  = lane >> 4;
  const int row  = lane & 15;

  // ---- stage packed weights to LDS with async DMA (ASYNCcnt path) ----
  {
    const unsigned lbase = (unsigned)(uintptr_t)&ldsB[0];
#pragma unroll
    for (int it = 0; it < PKOM_ELEMS * 2 / (128 * 16); ++it) {   // 18 chunks/thread
      const unsigned byteoff = (unsigned)(it * 128 + tid) * 16u;
      async_g2l_b128(lbase + byteoff, byteoff, (const void*)pk);
    }
    wait_async0();
    __syncthreads();
  }

  const int p0   = blockIdx.x * 64 + wave * 16;   // 16 consecutive pixels, same (b,y)
  const int b     = p0 >> 14;
  const int y     = (p0 >> 7) & (HO_ - 1);
  const int xbase = p0 & (WO_ - 1);
  const int xcol  = xbase + row;

  v8f acc0 = {}; v8f acc1 = {};
#pragma unroll
  for (int s = 0; s < NSTEP; ++s) {
    const int kk   = s >> 1;
    const int coff = (s & 1) * 32;
    const int c1   = coff + hlf * 16;
    const int yy = y    + kk / 3 - 1;
    const int xx = xcol + kk % 3 - 1;
    v16bf a;
    if (yy >= 0 && yy < H_ && xx >= 0 && xx < W_) {
      const float* base = xt + (((b * H_ + yy) * W_ + xx) << 6) + c1;
      v4f f0 = *(const v4f*)(base);
      v4f f1 = *(const v4f*)(base + 4);
      v4f f2 = *(const v4f*)(base + 8);
      v4f f3 = *(const v4f*)(base + 12);
#pragma unroll
      for (int i = 0; i < 4; ++i) {
        a[i]      = (__bf16)f0[i];
        a[4 + i]  = (__bf16)f1[i];
        a[8 + i]  = (__bf16)f2[i];
        a[12 + i] = (__bf16)f3[i];
      }
    } else {
#pragma unroll
      for (int i = 0; i < 16; ++i) a[i] = (__bf16)0.f;
    }
    const __bf16* pb = ldsB + ((s * 2) * 32 + lane) * 16;
    const v16bf b0 = *(const v16bf*)(pb);
    const v16bf b1 = *(const v16bf*)(pb + 32 * 16);
    acc0 = __builtin_amdgcn_wmma_f32_16x16x32_bf16(false, a, false, b0, (short)0, acc0, false, false);
    acc1 = __builtin_amdgcn_wmma_f32_16x16x32_bf16(false, a, false, b1, (short)0, acc1, false, false);
  }

  // epilogue: route channels -> dy / dx / sigmoid(mask), planar [B][KK][HO][WO]
#pragma unroll
  for (int n = 0; n < 2; ++n) {
    const int ch = n * 16 + row;              // D column = lane&15
    if (ch >= 27) continue;
    const v8f accv = n ? acc1 : acc0;
    const float bias = b_om[ch];
    float* dst; int kk2; bool sig;
    if (ch < 9)       { dst = dyA; kk2 = ch;      sig = false; }
    else if (ch < 18) { dst = dxA; kk2 = ch - 9;  sig = false; }
    else              { dst = mA;  kk2 = ch - 18; sig = true;  }
    const int rowbase = ((b * KK_ + kk2) * HO_ + y) * WO_ + xbase + hlf * 8;
#pragma unroll
    for (int v = 0; v < 8; ++v) {
      float val = accv[v] + bias;
      if (sig) val = 1.f / (1.f + __expf(-val));
      dst[rowbase + v] = val;
    }
  }
}

// ---------------- kernel 3: fused bilinear sampling + main GEMM ----------------
__global__ __launch_bounds__(128) void k_main(
    const float* __restrict__ xt, const __bf16* __restrict__ pk,
    const float* __restrict__ dyA, const float* __restrict__ dxA,
    const float* __restrict__ mA,
    const float* __restrict__ bias, float* __restrict__ out) {
  __shared__ __bf16 ldsB[PKW_ELEMS];

  const int tid  = threadIdx.x;
  const int lane = tid & 31;
  const int wave = tid >> 5;
  const int hlf  = lane >> 4;
  const int row  = lane & 15;

  // ---- stage packed weights to LDS with async DMA (ASYNCcnt path) ----
  {
    const unsigned lbase = (unsigned)(uintptr_t)&ldsB[0];
#pragma unroll
    for (int it = 0; it < PKW_ELEMS * 2 / (128 * 16); ++it) {    // 36 chunks/thread
      const unsigned byteoff = (unsigned)(it * 128 + tid) * 16u;
      async_g2l_b128(lbase + byteoff, byteoff, (const void*)pk);
    }
    wait_async0();
    __syncthreads();
  }

  const int p0   = blockIdx.x * 64 + wave * 16;
  const int b     = p0 >> 14;
  const int y     = (p0 >> 7) & (HO_ - 1);
  const int xbase = p0 & (WO_ - 1);
  const int xcol  = xbase + row;

  float dyv[9], dxv[9], mv[9];
#pragma unroll
  for (int kk = 0; kk < 9; ++kk) {
    const int idx = ((b * KK_ + kk) * HO_ + y) * WO_ + xcol;
    dyv[kk] = dyA[idx]; dxv[kk] = dxA[idx]; mv[kk] = mA[idx];
  }

  v8f acc0 = {}, acc1 = {}, acc2 = {}, acc3 = {};
#pragma unroll
  for (int s = 0; s < NSTEP; ++s) {
    const int kk   = s >> 1;
    const int coff = (s & 1) * 32;
    const int c1   = coff + hlf * 16;

    const float py = (float)(y - 1 + kk / 3) + dyv[kk];
    const float px = (float)(xcol - 1 + kk % 3) + dxv[kk];
    const float fy = floorf(py), fx = floorf(px);
    const int y0 = (int)fy, x0 = (int)fx;
    const float wy = py - fy, wx = px - fx;
    const float mval = mv[kk];
    const float wgt[4] = { (1.f - wy) * (1.f - wx) * mval,
                           (1.f - wy) * wx        * mval,
                           wy        * (1.f - wx) * mval,
                           wy        * wx         * mval };
    float sum[16];
#pragma unroll
    for (int i = 0; i < 16; ++i) sum[i] = 0.f;
#pragma unroll
    for (int j = 0; j < 4; ++j) {
      const int yi = y0 + (j >> 1);
      const int xi = x0 + (j & 1);
      const bool valid = (yi >= 0) && (yi < H_) && (xi >= 0) && (xi < W_);
      const float wj = valid ? wgt[j] : 0.f;
      const int yc = min(max(yi, 0), H_ - 1);
      const int xc = min(max(xi, 0), W_ - 1);
      const float* base = xt + (((b * H_ + yc) * W_ + xc) << 6) + c1;
      v4f f0 = *(const v4f*)(base);
      v4f f1 = *(const v4f*)(base + 4);
      v4f f2 = *(const v4f*)(base + 8);
      v4f f3 = *(const v4f*)(base + 12);
#pragma unroll
      for (int i = 0; i < 4; ++i) {
        sum[i]      += wj * f0[i];
        sum[4 + i]  += wj * f1[i];
        sum[8 + i]  += wj * f2[i];
        sum[12 + i] += wj * f3[i];
      }
    }
    v16bf a;
#pragma unroll
    for (int i = 0; i < 16; ++i) a[i] = (__bf16)sum[i];

    const __bf16* pb = ldsB + ((s * 4) * 32 + lane) * 16;
    const v16bf b0 = *(const v16bf*)(pb);
    const v16bf b1 = *(const v16bf*)(pb + 1 * 32 * 16);
    const v16bf b2 = *(const v16bf*)(pb + 2 * 32 * 16);
    const v16bf b3 = *(const v16bf*)(pb + 3 * 32 * 16);
    acc0 = __builtin_amdgcn_wmma_f32_16x16x32_bf16(false, a, false, b0, (short)0, acc0, false, false);
    acc1 = __builtin_amdgcn_wmma_f32_16x16x32_bf16(false, a, false, b1, (short)0, acc1, false, false);
    acc2 = __builtin_amdgcn_wmma_f32_16x16x32_bf16(false, a, false, b2, (short)0, acc2, false, false);
    acc3 = __builtin_amdgcn_wmma_f32_16x16x32_bf16(false, a, false, b3, (short)0, acc3, false, false);
  }

  // epilogue: NCHW fp32 output + bias
#pragma unroll
  for (int n = 0; n < 4; ++n) {
    const v8f accv = (n == 0) ? acc0 : (n == 1) ? acc1 : (n == 2) ? acc2 : acc3;
    const int co = n * 16 + row;
    const float bv = bias[co];
    float* dst = out + ((b * CO_ + co) * HO_ + y) * WO_ + xbase + hlf * 8;
#pragma unroll
    for (int v = 0; v < 8; ++v) dst[v] = accv[v] + bv;
  }
}

extern "C" void kernel_launch(void* const* d_in, const int* in_sizes, int n_in,
                              void* d_out, int out_size, void* d_ws, size_t ws_size,
                              hipStream_t stream) {
  const float* x    = (const float*)d_in[0];
  const float* w_om = (const float*)d_in[1];
  const float* b_om = (const float*)d_in[2];
  const float* w    = (const float*)d_in[3];
  const float* bias = (const float*)d_in[4];
  float* out = (float*)d_out;
  char*  ws  = (char*)d_ws;

  float*  xt   = (float*)(ws + OFF_XT);
  __bf16* pkOM = (__bf16*)(ws + OFF_PKOM);
  __bf16* pkW  = (__bf16*)(ws + OFF_PKW);
  float*  dyA  = (float*)(ws + OFF_DY);
  float*  dxA  = (float*)(ws + OFF_DX);
  float*  mA   = (float*)(ws + OFF_M);

  k_xt<<<(B_ * C_ * H_ * W_) / 256, 256, 0, stream>>>(x, xt);
  k_pack_om<<<(PKOM_ELEMS + 255) / 256, 256, 0, stream>>>(w_om, pkOM);
  k_pack_w<<<(PKW_ELEMS + 255) / 256, 256, 0, stream>>>(w, pkW);
  k_offset<<<(B_ * HO_ * WO_) / 64, 128, 0, stream>>>(xt, pkOM, b_om, dyA, dxA, mA);
  k_main<<<(B_ * HO_ * WO_) / 64, 128, 0, stream>>>(xt, pkW, dyA, dxA, mA, bias, out);
}
